// PerPixelCrossAttention_53085795779257
// MI455X (gfx1250) — matrix operand order repaired and checked by reading, not compiled, not run on hardware
//
#include <hip/hip_runtime.h>

// ---------------------------------------------------------------------------
// Per-pixel cross attention, bf16 WMMA path for gfx1250 (MI455X).
// B=4 HW=4096 L=32 D=128 H=2 DH=64 inner=128.
// d_ws layout (bf16 fragment-major weights, written by prep kernel):
//   [0      .. 16384) : Wq  frags (4 ktiles x  8 ntiles) * 512 bf16
//   [16384  .. 49152) : Wkv frags (4 ktiles x 16 ntiles) * 512 bf16
//   [49152  .. 65536) : Wo  frags (4 ktiles x  8 ntiles) * 512 bf16
// Requires ws_size >= 131072 bytes.
// ---------------------------------------------------------------------------

typedef __bf16 bf16;
typedef __attribute__((ext_vector_type(16))) __bf16 v16bf;
typedef __attribute__((ext_vector_type(8)))  __bf16 v8bf;
typedef __attribute__((ext_vector_type(8)))  float  v8f;

#define D_DIM 128
#define L_DIM 32
#define INNER 128
#define KVW   256
#define PIX   8            // pixels per workgroup
#define NPIX  16384        // B*HW
#define CP    136          // bf16 row pitch (272B, 16B-multiple, bank-skewed)
#define KVP   268          // f32 row pitch for KV scratch
#define QP    132          // f32 row pitch for Q scratch
#define SCALE 0.125f       // 64^-0.5

// ---- prep: shuffle fp32 weights into bf16 WMMA B-fragment layout ----------
// B operand (32x16 K*N tile), wave32: lane = 16*(k>=16half) + n; element e:
//   n = ntile*16 + (lane & 15),  k = ktile*32 + (lane>>4)*16 + e
__global__ void ppca_prep_weights(const float* __restrict__ Wq,
                                  const float* __restrict__ Wkv,
                                  const float* __restrict__ Wo,
                                  bf16* __restrict__ wqF,
                                  bf16* __restrict__ wkvF,
                                  bf16* __restrict__ woF) {
  int tid = blockIdx.x * 256 + threadIdx.x;   // 0 .. 65535
  if (tid < 16384) {                           // Wq: [128][128], 8 ntiles
    int f = tid >> 9, r = tid & 511, lane = r >> 4, e = r & 15;
    int nt = f & 7, kt = f >> 3;
    int n = nt * 16 + (lane & 15);
    int k = kt * 32 + (lane >> 4) * 16 + e;
    wqF[tid] = (bf16)Wq[k * INNER + n];
  } else if (tid < 49152) {                    // Wkv: [128][256], 16 ntiles
    int u = tid - 16384;
    int f = u >> 9, r = u & 511, lane = r >> 4, e = r & 15;
    int nt = f & 15, kt = f >> 4;
    int n = nt * 16 + (lane & 15);
    int k = kt * 32 + (lane >> 4) * 16 + e;
    wkvF[u] = (bf16)Wkv[k * KVW + n];
  } else {                                     // Wo: [128][128], 8 ntiles
    int u = tid - 49152;
    int f = u >> 9, r = u & 511, lane = r >> 4, e = r & 15;
    int nt = f & 7, kt = f >> 3;
    int n = nt * 16 + (lane & 15);
    int k = kt * 32 + (lane >> 4) * 16 + e;
    woF[u] = (bf16)Wo[k * D_DIM + n];
  }
}

// ---------------------------------------------------------------------------
__device__ __forceinline__ v8f wmma_bf16(v16bf a, v16bf b, v8f c) {
  // V_WMMA_F32_16X16X32_BF16: D = A(16x32) x B(32x16) + C
  return __builtin_amdgcn_wmma_f32_16x16x32_bf16(
      /*neg_a=*/false, a, /*neg_b=*/false, b,
      /*c_mod=*/(short)0, c, /*reuse_a=*/false, /*reuse_b=*/false);
}

// A operand from row-major bf16 LDS tile (contiguous k order, same as the prepped B frags):
//   row = m0 + (lane&15); lanes<16: e0-15 -> K k0+0..15
//                         lanes>=16: e0-15 -> K k0+16..31
__device__ __forceinline__ v16bf load_a_frag(const bf16* lds, int pitch,
                                             int m0, int k0, int lane) {
  int row = m0 + (lane & 15);
  int h   = lane >> 4;
  const bf16* p = lds + row * pitch + k0 + h * 16;
  union { v16bf v; v8bf x[2]; } u;
  u.x[0] = *(const v8bf*)(p);        // ds_load_b128
  u.x[1] = *(const v8bf*)(p + 8);    // ds_load_b128
  return u.v;
}

// ---------------------------------------------------------------------------
__global__ __launch_bounds__(128) void ppca_main(
    const float* __restrict__ x, const float* __restrict__ ctx,
    const float* __restrict__ bo,
    const bf16* __restrict__ wqF, const bf16* __restrict__ wkvF,
    const bf16* __restrict__ woF, float* __restrict__ out) {

  __shared__ __align__(32) bf16  Xb[16 * CP];          // 16x128 x-tile (bf16)
  __shared__ __align__(32) bf16  Ob[16 * CP];          // 16x128 attn-out tile
  __shared__ __align__(32) bf16  Cb[2][L_DIM * CP];    // double-buffered ctx
  __shared__ float KV[L_DIM * KVP];                    // 32x256 fp32 KV
  __shared__ float Qs[16 * QP];                        // Q rows (fp32)
  __shared__ float simS[2][L_DIM];
  __shared__ float attnS[2][L_DIM];

  const int t = threadIdx.x, lane = t & 31, w = t >> 5; // 4 waves
  const int P0 = blockIdx.x * PIX;                      // first pixel row
  const v16bf* wqV  = (const v16bf*)wqF;
  const v16bf* wkvV = (const v16bf*)wkvF;
  const v16bf* woV  = (const v16bf*)woF;

  // zero X/O tiles (pad rows 8..15 and pad cols stay 0)
  for (int i = t; i < 16 * CP; i += 128) { Xb[i] = (bf16)0.0f; Ob[i] = (bf16)0.0f; }
  __syncthreads();
  for (int i = t; i < PIX * D_DIM; i += 128) {
    int r = i >> 7, c = i & 127;
    Xb[r * CP + c] = (bf16)x[(size_t)(P0 + r) * D_DIM + c];
  }
  __syncthreads();

  // ---- Q = X @ Wq  (one 16x128 M-tile over 8 pixels) ----------------------
  v8f qacc[2] = {};
  #pragma unroll
  for (int k = 0; k < 4; ++k) {
    v16bf a = load_a_frag(Xb, CP, 0, k * 32, lane);
    #pragma unroll
    for (int n = 0; n < 2; ++n) {
      v16bf bm = wqV[(k * 8 + (w * 2 + n)) * 32 + lane];
      qacc[n] = wmma_bf16(a, bm, qacc[n]);
    }
  }
  #pragma unroll
  for (int n = 0; n < 2; ++n) {
    int col = (w * 2 + n) * 16 + (lane & 15);
    #pragma unroll
    for (int j = 0; j < 8; ++j) {
      int row = (lane >> 4) * 8 + j;                 // D-tile layout
      Qs[row * QP + col] = qacc[n][j];
    }
  }

  // stage context of first pixel into Cb[0] (fp32 -> bf16)
  {
    const float4* s4 = (const float4*)(ctx + (size_t)P0 * (L_DIM * D_DIM));
    for (int j = t; j < (L_DIM * D_DIM) / 4; j += 128) {
      float4 v = s4[j];
      int r = j >> 5, c = (j & 31) * 4;
      bf16* d = &Cb[0][r * CP + c];
      d[0] = (bf16)v.x; d[1] = (bf16)v.y; d[2] = (bf16)v.z; d[3] = (bf16)v.w;
    }
  }
  __syncthreads();

  for (int p = 0; p < PIX; ++p) {
    const int buf = p & 1;
    if (p + 1 < PIX)   // warm L2/L1 for next pixel's context (global_prefetch_b8)
      __builtin_prefetch(ctx + (size_t)(P0 + p + 1) * (L_DIM * D_DIM) + t * 32, 0, 0);

    // ---- KV = C @ Wkv  (32x128 @ 128x256), wave w owns n-tiles 4w..4w+3 ----
    v8f acc[2][4] = {};
    #pragma unroll
    for (int k = 0; k < 4; ++k) {
      v16bf a0 = load_a_frag(Cb[buf], CP, 0,  k * 32, lane);
      v16bf a1 = load_a_frag(Cb[buf], CP, 16, k * 32, lane);
      #pragma unroll
      for (int n = 0; n < 4; ++n) {
        v16bf bm = wkvV[(k * 16 + (w * 4 + n)) * 32 + lane];
        acc[0][n] = wmma_bf16(a0, bm, acc[0][n]);
        acc[1][n] = wmma_bf16(a1, bm, acc[1][n]);
      }
    }
    // spill KV tiles to LDS (fp32, D-tile layout)
    #pragma unroll
    for (int mi = 0; mi < 2; ++mi) {
      #pragma unroll
      for (int n = 0; n < 4; ++n) {
        int col = (w * 4 + n) * 16 + (lane & 15);
        #pragma unroll
        for (int j = 0; j < 8; ++j) {
          int row = mi * 16 + (lane >> 4) * 8 + j;
          KV[row * KVP + col] = acc[mi][n][j];
        }
      }
    }
    __syncthreads();

    // stage next pixel's context into the other buffer (overlaps attention)
    if (p + 1 < PIX) {
      const float4* s4 = (const float4*)(ctx + (size_t)(P0 + p + 1) * (L_DIM * D_DIM));
      for (int j = t; j < (L_DIM * D_DIM) / 4; j += 128) {
        float4 v = s4[j];
        int r = j >> 5, c = (j & 31) * 4;
        bf16* d = &Cb[1 - buf][r * CP + c];
        d[0] = (bf16)v.x; d[1] = (bf16)v.y; d[2] = (bf16)v.z; d[3] = (bf16)v.w;
      }
    }

    // ---- tiny per-pixel attention: sim -> softmax -> attn@V (VALU) --------
    if (t < 64) {                       // (h, l): 2 heads x 32 keys
      int h = t >> 5, l = t & 31;
      float s = 0.f;
      #pragma unroll 8
      for (int d = 0; d < 64; ++d)
        s += Qs[p * QP + h * 64 + d] * KV[l * KVP + h * 64 + d];
      simS[h][l] = s * SCALE;
    }
    __syncthreads();
    if (t < 64) {
      int h = t >> 5, l = t & 31;
      float m = -3.4e38f;
      for (int j = 0; j < 32; ++j) m = fmaxf(m, simS[h][j]);
      float sum = 0.f;
      for (int j = 0; j < 32; ++j) sum += __expf(simS[h][j] - m);
      attnS[h][l] = __expf(simS[h][l] - m) / sum;
    }
    __syncthreads();
    {
      int h = t >> 6, d = t & 63;       // 128 threads: (h, d)
      float o = 0.f;
      #pragma unroll 8
      for (int l = 0; l < 32; ++l)
        o += attnS[h][l] * KV[l * KVP + 128 + h * 64 + d];
      Ob[p * CP + h * 64 + d] = (bf16)o;
    }
    __syncthreads();
  }

  // ---- out = O @ Wo + bo  (one 16x128 M-tile over 8 pixels) ---------------
  v8f oacc[2] = {};
  #pragma unroll
  for (int k = 0; k < 4; ++k) {
    v16bf a = load_a_frag(Ob, CP, 0, k * 32, lane);
    #pragma unroll
    for (int n = 0; n < 2; ++n) {
      v16bf bm = woV[(k * 8 + (w * 2 + n)) * 32 + lane];
      oacc[n] = wmma_bf16(a, bm, oacc[n]);
    }
  }
  #pragma unroll
  for (int n = 0; n < 2; ++n) {
    int col = (w * 2 + n) * 16 + (lane & 15);
    float bias = bo[col];
    #pragma unroll
    for (int j = 0; j < 8; ++j) {
      int row = (lane >> 4) * 8 + j;
      if (row < PIX)
        out[(size_t)(P0 + row) * D_DIM + col] = oacc[n][j] + bias;
    }
  }
}

// ---------------------------------------------------------------------------
extern "C" void kernel_launch(void* const* d_in, const int* in_sizes, int n_in,
                              void* d_out, int out_size, void* d_ws, size_t ws_size,
                              hipStream_t stream) {
  const float* x   = (const float*)d_in[0];
  const float* ctx = (const float*)d_in[1];
  const float* Wq  = (const float*)d_in[2];
  const float* Wkv = (const float*)d_in[3];
  const float* Wo  = (const float*)d_in[4];
  const float* bo  = (const float*)d_in[5];
  float* out = (float*)d_out;

  bf16* wqF  = (bf16*)d_ws;           // 16384 bf16
  bf16* wkvF = wqF + 16384;           // 32768 bf16
  bf16* woF  = wkvF + 32768;          // 16384 bf16

  ppca_prep_weights<<<256, 256, 0, stream>>>(Wq, Wkv, Wo, wqF, wkvF, woF);
  ppca_main<<<NPIX / PIX, 128, 0, stream>>>(x, ctx, bo, wqF, wkvF, woF, out);
}
